// PANLoss_19267223290039
// MI455X (gfx1250) — matrix operand order repaired and checked by reading, not compiled, not run on hardware
//
#include <hip/hip_runtime.h>
#include <cstddef>

typedef __attribute__((ext_vector_type(16))) _Float16 v16h;
typedef __attribute__((ext_vector_type(8)))  float    v8f;

#define HWPIX (512*512)

// ---- order-preserving float <-> u32 key (ascending) ----
__device__ __forceinline__ unsigned fkey(float f){
  unsigned u = __float_as_uint(f);
  return (u & 0x80000000u) ? ~u : (u | 0x80000000u);
}
__device__ __forceinline__ float fkey_inv(unsigned k){
  unsigned u = (k & 0x80000000u) ? (k & 0x7fffffffu) : ~k;
  return __uint_as_float(u);
}

// hardware-rate sigmoid: v_exp_f32 + v_rcp_f32 instead of IEEE div sequence
__device__ __forceinline__ float fast_sigmoid(float x){
  return __builtin_amdgcn_rcpf(1.f + __expf(-x));
}

__device__ __forceinline__ float blockReduceF(float v, float* buf){
  int t = threadIdx.x; buf[t]=v; __syncthreads();
  for(int s=128;s>0;s>>=1){ if(t<s) buf[t]+=buf[t+s]; __syncthreads(); }
  float r = buf[0]; __syncthreads(); return r;
}
__device__ __forceinline__ int blockReduceI(int v, int* buf){
  int t = threadIdx.x; buf[t]=v; __syncthreads();
  for(int s=128;s>0;s>>=1){ if(t<s) buf[t]+=buf[t+s]; __syncthreads(); }
  int r = buf[0]; __syncthreads(); return r;
}

__global__ void pan_zero(unsigned* p, size_t nwords){
  size_t i = (size_t)blockIdx.x*blockDim.x + threadIdx.x;
  const size_t stride = (size_t)gridDim.x*blockDim.x;
  for(; i<nwords; i+=stride) p[i]=0u;
}

// ---------------------------------------------------------------------------
// Pass 1: WMMA mask-matmul segment sums (kcnt/ksum/tcnt), pos/neg counts,
//         kernel-dice partial sums, hi-16 histogram of negative scores.
// Grid: (32, N) blocks of 256 threads (8 waves). Each wave: 32 tiles of 32 px.
// WMMA mapping per 32-px tile:
//   A[16x32] f16: rows 0..7 = (kernel==i+1), rows 8..15 = (text==i+1)  (exact)
//   B[32x16] f16: cols 0..3 = embed ch, col 4 = 1.0 (counts), col 5 = lg (pass2)
//   C[16x16] f32 accumulates over all tiles of the wave.
// ---------------------------------------------------------------------------
__global__ __launch_bounds__(256) void pan_pass1(
    const float* __restrict__ preds, const int* __restrict__ gtext,
    const int* __restrict__ gkern, const int* __restrict__ gmask,
    float* __restrict__ acc1, int* __restrict__ ipos, int* __restrict__ ineg,
    float* __restrict__ diceK, unsigned* __restrict__ histHi)
{
  const int n    = blockIdx.y;
  const int wave = threadIdx.x >> 5;
  const int lane = threadIdx.x & 31;
  const int m    = lane & 15;          // A: row M (both halves); B/C: column N
  const int hi   = lane >> 4;          // K half (A/B), +8 row offset (C)
  const int waveG = blockIdx.x*8 + wave;

  const float* ptext = preds + ((size_t)n*6 + 0)*HWPIX;
  const float* pkern = preds + ((size_t)n*6 + 1)*HWPIX;
  const float* pemb  = preds + ((size_t)n*6 + 2)*HWPIX;
  const int* tl = gtext + (size_t)n*HWPIX;
  const int* kl = gkern + (size_t)n*HWPIX;
  const int* ml = gmask + (size_t)n*HWPIX;
  unsigned* hh = histHi + (size_t)n*65536;

  __shared__ int      sK[8][32], sT[8][32];
  __shared__ _Float16 sE[8][4][32];
  __shared__ float    redf[256];
  __shared__ int      redi[256];

  v8f cacc = {};
  int posA=0, negA=0; float aK=0.f,bK=0.f,cK=0.f;

  for(int it=0; it<32; ++it){
    const int p = (waveG*32 + it)*32 + lane;
    const int t = tl[p], k = kl[p], mm = ml[p];
    const float e0 = pemb[p];
    const float e1 = pemb[(size_t)HWPIX + p];
    const float e2 = pemb[2*(size_t)HWPIX + p];
    const float e3 = pemb[3*(size_t)HWPIX + p];
    const float sc = ptext[p];
    const bool pos = (t>0), mp = (mm>0);
    posA += (pos && mp) ? 1 : 0;
    negA += pos ? 0 : 1;
    if(!pos) atomicAdd(&hh[fkey(sc)>>16], 1u);
    {
      const float w  = (pos&&mp)?1.f:0.f;         // samp_k
      const float s  = fast_sigmoid(pkern[p]);
      const float pv = s*w, tv = (k>0)? w : 0.f;
      aK += pv*tv; bK += pv*pv; cK += tv*tv;
    }
    sK[wave][lane]=k; sT[wave][lane]=t;
    sE[wave][0][lane]=(_Float16)e0; sE[wave][1][lane]=(_Float16)e1;
    sE[wave][2][lane]=(_Float16)e2; sE[wave][3][lane]=(_Float16)e3;
    asm volatile("s_wait_dscnt 0" ::: "memory");

    v16h A, B;
    const int tgt = (m&7)+1;
    #pragma unroll
    for(int v=0; v<8; ++v){                 // A 16x32 layout (ISA 7.12.2)
      const int kb = ((v<4)? 2*v : 16+2*(v-4)) + 8*hi;
      const int l0 = (m<8)? sK[wave][kb]   : sT[wave][kb];
      const int l1 = (m<8)? sK[wave][kb+1] : sT[wave][kb+1];
      A[2*v]   = (l0==tgt)?(_Float16)1.0f:(_Float16)0.0f;
      A[2*v+1] = (l1==tgt)?(_Float16)1.0f:(_Float16)0.0f;
    }
    #pragma unroll
    for(int v=0; v<8; ++v){                 // B 32x16, K split by lane half
      const int kk = ((v<4)? 2*v : 16+2*(v-4)) + 8*hi;
      _Float16 b0, b1;
      if (m<4){ b0=sE[wave][m][kk]; b1=sE[wave][m][kk+1]; }
      else if (m==4){ b0=(_Float16)1.0f; b1=(_Float16)1.0f; }
      else { b0=(_Float16)0.0f; b1=(_Float16)0.0f; }
      B[2*v]=b0; B[2*v+1]=b1;
    }
    cacc = __builtin_amdgcn_wmma_f32_16x16x32_f16(false, A, false, B,
                                                  (short)0, cacc, false, false);
  }
  if (m < 6){                               // C: VGPR r holds M=r (+8 if hi)
    float* ab = acc1 + (size_t)n*96;
    #pragma unroll
    for(int r=0;r<8;++r) atomicAdd(&ab[(r+8*hi)*6 + m], cacc[r]);
  }
  const int posS = blockReduceI(posA, redi);
  const int negS = blockReduceI(negA, redi);
  const float aS = blockReduceF(aK, redf);
  const float bS = blockReduceF(bK, redf);
  const float cS = blockReduceF(cK, redf);
  if(threadIdx.x==0){
    atomicAdd(&ipos[n], posS); atomicAdd(&ineg[n], negS);
    atomicAdd(&diceK[n*3+0], aS); atomicAdd(&diceK[n*3+1], bS); atomicAdd(&diceK[n*3+2], cS);
  }
}

// ---------------------------------------------------------------------------
// Prep: per-batch avgs/valid from acc1; neg_num; select hi-16 bucket of the
// neg_num-th largest negative score (descending radix select, stage 1).
// ---------------------------------------------------------------------------
__global__ __launch_bounds__(256) void pan_prep(
    const unsigned* __restrict__ histHi, const float* __restrict__ acc1,
    const int* __restrict__ ipos, const int* __restrict__ ineg,
    float* __restrict__ avgs, float* __restrict__ validArr,
    int* __restrict__ selhi, int* __restrict__ remrank, int* __restrict__ fb)
{
  const int n = blockIdx.x;
  __shared__ unsigned S[256];
  __shared__ int sFb, sR, sJ, sRem;
  const unsigned* h = histHi + (size_t)n*65536;
  const int base = 65536 - 256*((int)threadIdx.x+1);   // chunk 0 = top bins
  unsigned s=0;
  for(int b=0;b<256;++b) s += h[base+b];
  S[threadIdx.x]=s;
  if(threadIdx.x==0){
    const int pos = ipos[n]; const int negc = ineg[n];
    int r = pos*3; if(r>negc) r=negc;
    const int f = (pos==0 || r==0) ? 1 : 0;
    sFb=f; sR=r; fb[n]=f;
    if(f){ selhi[n]=0; remrank[n]=0; }
  }
  __syncthreads();
  if(threadIdx.x==0){
    if(!sFb){
      unsigned cum=0; int js=0, rem=sR;
      for(int j=0;j<256;++j){
        if(cum+S[j] >= (unsigned)sR){ js=j; rem=sR-(int)cum; break; }
        cum+=S[j];
      }
      sJ=js; sRem=rem;
    } else sJ=-1;
  }
  __syncthreads();
  if((int)threadIdx.x==sJ){
    const int b0 = 65536 - 256*(sJ+1);
    unsigned cum=0; const int r=sRem;
    for(int b=255;b>=0;--b){
      const unsigned c=h[b0+b];
      if(cum+c >= (unsigned)r){ selhi[n]=b0+b; remrank[n]=r-(int)cum; break; }
      cum+=c;
    }
  }
  if(threadIdx.x < 8){
    const int i = threadIdx.x;
    const float* A = acc1 + (size_t)n*96;
    const float kc = A[i*6+4], tc = A[(8+i)*6+4];
    const float inv = 1.f/fmaxf(kc,1.f);
    for(int c=0;c<4;++c) avgs[((size_t)n*8+i)*4+c] = A[i*6+c]*inv;
    validArr[(size_t)n*8+i] = (kc>0.f && tc>0.f)?1.f:0.f;
  }
}

// Radix select stage 2a: histogram low 16 bits within the chosen hi bucket.
__global__ __launch_bounds__(256) void pan_histlo(
    const float* __restrict__ preds, const int* __restrict__ gtext,
    const int* __restrict__ selhi, const int* __restrict__ fb,
    unsigned* __restrict__ histLo)
{
  const int n = blockIdx.y;
  if(fb[n]) return;
  const unsigned sh = (unsigned)selhi[n];
  const float* sc = preds + ((size_t)n*6)*HWPIX;
  const int* tl = gtext + (size_t)n*HWPIX;
  unsigned* hl = histLo + (size_t)n*65536;
  const int base = blockIdx.x * (256*16);
  for(int u=0;u<16;++u){
    const int p = base + u*256 + threadIdx.x;
    if(tl[p]==0){
      const unsigned key = fkey(sc[p]);
      if((key>>16)==sh) atomicAdd(&hl[key & 0xffffu], 1u);
    }
  }
}

// Radix select stage 2b: scan lo histogram -> exact OHEM threshold value.
__global__ __launch_bounds__(256) void pan_pickthr(
    const unsigned* __restrict__ histLo, const int* __restrict__ selhi,
    const int* __restrict__ remrank, const int* __restrict__ fb,
    float* __restrict__ thr)
{
  const int n = blockIdx.x;
  if(fb[n]){ if(threadIdx.x==0) thr[n]=0.f; return; }
  __shared__ unsigned S[256];
  __shared__ int sJ, sRem;
  const unsigned* h = histLo + (size_t)n*65536;
  const int base = 65536 - 256*((int)threadIdx.x+1);
  unsigned s=0;
  for(int b=0;b<256;++b) s += h[base+b];
  S[threadIdx.x]=s;
  __syncthreads();
  if(threadIdx.x==0){
    const int r = remrank[n];
    unsigned cum=0; int js=0, rem=r;
    for(int j=0;j<256;++j){
      if(cum+S[j] >= (unsigned)r){ js=j; rem=r-(int)cum; break; }
      cum+=S[j];
    }
    sJ=js; sRem=rem;
  }
  __syncthreads();
  if((int)threadIdx.x==sJ){
    const int b0 = 65536 - 256*(sJ+1);
    unsigned cum=0; const int r=sRem;
    for(int b=255;b>=0;--b){
      const unsigned c=h[b0+b];
      if(cum+c >= (unsigned)r){
        const unsigned key = ((unsigned)selhi[n]<<16) | (unsigned)(b0+b);
        thr[n] = fkey_inv(key);
        break;
      }
      cum+=c;
    }
  }
}

// ---------------------------------------------------------------------------
// Pass 2: per-pixel lg (needs avgs), WMMA aggregation numerators (tmask @ lg),
//         text-dice sums with OHEM-selected mask.
// ---------------------------------------------------------------------------
__global__ __launch_bounds__(256) void pan_pass2(
    const float* __restrict__ preds, const int* __restrict__ gtext,
    const int* __restrict__ gkern, const int* __restrict__ gmask,
    const float* __restrict__ avgs, const float* __restrict__ thrArr,
    const int* __restrict__ fbArr,
    float* __restrict__ acc2, float* __restrict__ diceT)
{
  const int n    = blockIdx.y;
  const int wave = threadIdx.x >> 5;
  const int lane = threadIdx.x & 31;
  const int m    = lane & 15;
  const int hi   = lane >> 4;
  const int waveG = blockIdx.x*8 + wave;

  const float* ptext = preds + ((size_t)n*6 + 0)*HWPIX;
  const float* pemb  = preds + ((size_t)n*6 + 2)*HWPIX;
  const int* tl = gtext + (size_t)n*HWPIX;
  const int* kl = gkern + (size_t)n*HWPIX;
  const int* ml = gmask + (size_t)n*HWPIX;

  __shared__ float sAvg[9][4];
  __shared__ float sThr; __shared__ int sFb;
  __shared__ int      sK[8][32], sT[8][32];
  __shared__ _Float16 sE[8][4][32];
  __shared__ _Float16 sL[8][32];
  __shared__ float    redf[256];

  if(threadIdx.x < 36){
    const int i=threadIdx.x>>2, c=threadIdx.x&3;
    sAvg[i][c] = (i==0)?0.f : avgs[((size_t)n*8 + (i-1))*4 + c];
  }
  if(threadIdx.x==0){ sThr = thrArr[n]; sFb = fbArr[n]; }
  __syncthreads();
  const float thr = sThr; const int fb = sFb;

  v8f cacc = {};
  float aT=0.f,bT=0.f,cT=0.f;

  for(int it=0; it<32; ++it){
    const int p = (waveG*32 + it)*32 + lane;
    int t = tl[p]; const int k = kl[p], mm = ml[p];
    if(t<0) t=0; if(t>8) t=8;
    const float e0 = pemb[p];
    const float e1 = pemb[(size_t)HWPIX + p];
    const float e2 = pemb[2*(size_t)HWPIX + p];
    const float e3 = pemb[3*(size_t)HWPIX + p];
    const float sc = ptext[p];
    const float d0=e0-sAvg[t][0], d1=e1-sAvg[t][1], d2=e2-sAvg[t][2], d3=e3-sAvg[t][3];
    const float dist = sqrtf(d0*d0+d1*d1+d2*d2+d3*d3 + 1e-12f) - 0.5f;
    const float dd = fmaxf(dist,0.f);
    const float lg = log1pf(dd*dd);
    const bool pos=(t>0), mp=(mm>0);
    const float samp = fb ? (mp?1.f:0.f)
                          : ((((sc>=thr)||pos)&&mp) ? 1.f : 0.f);
    const float s  = fast_sigmoid(sc);
    const float pv = s*samp, tv = pos? samp : 0.f;
    aT += pv*tv; bT += pv*pv; cT += tv*tv;

    sK[wave][lane]=k; sT[wave][lane]=t;
    sE[wave][0][lane]=(_Float16)e0; sE[wave][1][lane]=(_Float16)e1;
    sE[wave][2][lane]=(_Float16)e2; sE[wave][3][lane]=(_Float16)e3;
    sL[wave][lane]=(_Float16)lg;
    asm volatile("s_wait_dscnt 0" ::: "memory");

    v16h A, B;
    const int tgt = (m&7)+1;
    #pragma unroll
    for(int v=0; v<8; ++v){
      const int kb = ((v<4)? 2*v : 16+2*(v-4)) + 8*hi;
      const int l0 = (m<8)? sK[wave][kb]   : sT[wave][kb];
      const int l1 = (m<8)? sK[wave][kb+1] : sT[wave][kb+1];
      A[2*v]   = (l0==tgt)?(_Float16)1.0f:(_Float16)0.0f;
      A[2*v+1] = (l1==tgt)?(_Float16)1.0f:(_Float16)0.0f;
    }
    #pragma unroll
    for(int v=0; v<8; ++v){
      const int kk = ((v<4)? 2*v : 16+2*(v-4)) + 8*hi;
      _Float16 b0, b1;
      if (m<4){ b0=sE[wave][m][kk]; b1=sE[wave][m][kk+1]; }
      else if (m==4){ b0=(_Float16)1.0f; b1=(_Float16)1.0f; }
      else if (m==5){ b0=sL[wave][kk]; b1=sL[wave][kk+1]; }
      else { b0=(_Float16)0.0f; b1=(_Float16)0.0f; }
      B[2*v]=b0; B[2*v+1]=b1;
    }
    cacc = __builtin_amdgcn_wmma_f32_16x16x32_f16(false, A, false, B,
                                                  (short)0, cacc, false, false);
  }
  if (m < 6){
    float* ab = acc2 + (size_t)n*96;
    #pragma unroll
    for(int r=0;r<8;++r) atomicAdd(&ab[(r+8*hi)*6 + m], cacc[r]);
  }
  const float aS = blockReduceF(aT, redf);
  const float bS = blockReduceF(bT, redf);
  const float cS = blockReduceF(cT, redf);
  if(threadIdx.x==0){
    atomicAdd(&diceT[n*3+0], aS); atomicAdd(&diceT[n*3+1], bS); atomicAdd(&diceT[n*3+2], cS);
  }
}

// ---------------------------------------------------------------------------
// Finalize: per-batch losses, discrimination pairs, batch means + coefs.
// ---------------------------------------------------------------------------
__global__ void pan_final(const float* __restrict__ acc1, const float* __restrict__ acc2,
                          const float* __restrict__ diceK, const float* __restrict__ diceT,
                          const float* __restrict__ avgs, const float* __restrict__ validArr,
                          float* __restrict__ out, int N)
{
  __shared__ float L[64][4];
  const int n = threadIdx.x;
  if(n < N){
    const float* A1 = acc1 + (size_t)n*96;
    const float* A2 = acc2 + (size_t)n*96;
    const float* V  = validArr + (size_t)n*8;
    const float* AV = avgs + (size_t)n*32;
    float nv=0.f;
    for(int i=0;i<8;++i) nv += V[i];
    float aggr=0.f;
    for(int i=0;i<8;++i){
      const float tc = A1[(8+i)*6+4];
      aggr += V[i] * (A2[(8+i)*6+5] / fmaxf(tc,1.f));
    }
    aggr = (nv>0.f) ? aggr/fmaxf(nv,1.f) : 0.f;
    float discr=0.f;
    for(int i=0;i<8;++i) for(int j=i+1;j<8;++j){
      if(V[i]>0.f && V[j]>0.f){
        float sq=0.f;
        for(int c=0;c<4;++c){ const float d=AV[i*4+c]-AV[j*4+c]; sq+=d*d; }
        const float d = sqrtf(sq);
        const float hh = fmaxf(3.0f - d, 0.f);
        discr += log1pf(hh*hh);
      }
    }
    discr = (nv>1.f) ? discr / fmaxf(nv*(nv-1.f),1.f) : 0.f;
    const float aT=diceT[n*3+0]+1e-3f, bT=diceT[n*3+1]+1e-3f, cT=diceT[n*3+2]+1e-3f;
    const float aK=diceK[n*3+0]+1e-3f, bK=diceK[n*3+1]+1e-3f, cK=diceK[n*3+2]+1e-3f;
    L[n][0] = 1.f - 2.f*aT/(bT+cT);
    L[n][1] = 1.f - 2.f*aK/(bK+cK);
    L[n][2] = aggr;
    L[n][3] = discr;
  }
  __syncthreads();
  if(n==0){
    float s0=0.f,s1=0.f,s2=0.f,s3=0.f;
    for(int i=0;i<N;++i){ s0+=L[i][0]; s1+=L[i][1]; s2+=L[i][2]; s3+=L[i][3]; }
    const float invN = 1.f/(float)N;
    out[0]=s0*invN; out[1]=0.5f*s1*invN; out[2]=0.25f*s2*invN; out[3]=0.25f*s3*invN;
  }
}

extern "C" void kernel_launch(void* const* d_in, const int* in_sizes, int n_in,
                              void* d_out, int out_size, void* d_ws, size_t ws_size,
                              hipStream_t stream) {
  (void)n_in; (void)out_size; (void)ws_size;
  const float* preds = (const float*)d_in[0];
  const int* gtext = (const int*)d_in[1];
  const int* gkern = (const int*)d_in[2];
  const int* gmask = (const int*)d_in[3];
  const int N = in_sizes[1] / HWPIX;   // 16

  char* ws = (char*)d_ws;
  size_t off = 0;
  auto carve = [&](size_t bytes)->char*{
    char* p = ws + off;
    off += (bytes + 255) & ~(size_t)255;
    return p;
  };
  float*    acc1    = (float*)   carve((size_t)N*96*sizeof(float));
  float*    acc2    = (float*)   carve((size_t)N*96*sizeof(float));
  int*      ipos    = (int*)     carve((size_t)N*sizeof(int));
  int*      ineg    = (int*)     carve((size_t)N*sizeof(int));
  float*    diceK   = (float*)   carve((size_t)N*3*sizeof(float));
  float*    diceT   = (float*)   carve((size_t)N*3*sizeof(float));
  float*    avgs    = (float*)   carve((size_t)N*32*sizeof(float));
  float*    validA  = (float*)   carve((size_t)N*8*sizeof(float));
  int*      selhi   = (int*)     carve((size_t)N*sizeof(int));
  int*      remrank = (int*)     carve((size_t)N*sizeof(int));
  int*      fb      = (int*)     carve((size_t)N*sizeof(int));
  float*    thr     = (float*)   carve((size_t)N*sizeof(float));
  unsigned* histHi  = (unsigned*)carve((size_t)N*65536*sizeof(unsigned));
  unsigned* histLo  = (unsigned*)carve((size_t)N*65536*sizeof(unsigned));

  const size_t zeroWords = off / 4;   // off is 256B aligned
  pan_zero<<<2048, 256, 0, stream>>>((unsigned*)ws, zeroWords);

  pan_pass1<<<dim3(32, N), 256, 0, stream>>>(preds, gtext, gkern, gmask,
                                             acc1, ipos, ineg, diceK, histHi);
  pan_prep<<<N, 256, 0, stream>>>(histHi, acc1, ipos, ineg,
                                  avgs, validA, selhi, remrank, fb);
  pan_histlo<<<dim3(64, N), 256, 0, stream>>>(preds, gtext, selhi, fb, histLo);
  pan_pickthr<<<N, 256, 0, stream>>>(histLo, selhi, remrank, fb, thr);
  pan_pass2<<<dim3(32, N), 256, 0, stream>>>(preds, gtext, gkern, gmask,
                                             avgs, thr, fb, acc2, diceT);
  pan_final<<<1, 64, 0, stream>>>(acc1, acc2, diceK, diceT, avgs, validA,
                                  (float*)d_out, N);
}
